// VQ_VAE_88270167867709
// MI455X (gfx1250) — matrix operand order repaired and checked by reading, not compiled, not run on hardware
//
#include <hip/hip_runtime.h>
#include <cstdint>

typedef __attribute__((ext_vector_type(16))) _Float16 v16h;
typedef __attribute__((ext_vector_type(8)))  _Float16 v8h;
typedef __attribute__((ext_vector_type(8)))  float    v8f;
typedef __attribute__((ext_vector_type(4)))  int      v4i;

union V16 { v16h v; v8h h[2]; };
union G4  { uint32_t u[2]; _Float16 f[4]; };

__device__ __forceinline__ v8f wmma_f16(v16h a, v16h b, v8f c) {
  // D = A(16x32 f16) * B(32x16 f16) + C(16x16 f32)
  return __builtin_amdgcn_wmma_f32_16x16x32_f16(false, a, false, b, (short)0, c, false, false);
}

// ---- async global->LDS staging (gfx1250 ASYNCcnt path), with fallbacks -----
#if __has_builtin(__builtin_amdgcn_global_load_async_to_lds_b128)
#define ASYNC_MODE 1
#elif __has_builtin(__builtin_amdgcn_load_to_lds)
#define ASYNC_MODE 2
#pragma message("using __builtin_amdgcn_load_to_lds fallback for LDS staging")
#else
#define ASYNC_MODE 0
#pragma message("no async-to-LDS builtin available; using VGPR-bounce fallback")
#endif

typedef __attribute__((address_space(1))) v4i gv4i;   // global int4
typedef __attribute__((address_space(3))) v4i lv4i;   // LDS int4

// Copy `chunks16` x 16B from contiguous global `src` into LDS `dst`,
// cooperatively across 256 threads. Caller must __syncthreads() after.
__device__ __forceinline__ void stage_panel_256(const _Float16* __restrict__ src,
                                                _Float16* __restrict__ dst,
                                                int chunks16) {
#if ASYNC_MODE == 1
  {
    gv4i* gsrc = (gv4i*)(v4i*)(char*)src;
    lv4i* ldst = (lv4i*)(v4i*)(char*)dst;
    for (int c = threadIdx.x; c < chunks16; c += 256) {
      __builtin_amdgcn_global_load_async_to_lds_b128(gsrc + c, ldst + c, 0, 0);
    }
    asm volatile("s_wait_asynccnt 0x0" ::: "memory");
  }
#elif ASYNC_MODE == 2
  {
    auto gsrc = (char*)src;
    auto ldst = (__attribute__((address_space(3))) char*)(char*)dst;
    for (int c = threadIdx.x; c < chunks16; c += 256) {
      __builtin_amdgcn_load_to_lds(
          (void*)(gsrc + ((size_t)c << 4)),
          (__attribute__((address_space(3))) void*)(ldst + ((size_t)c << 4)),
          16, 0, 0);
    }
  }
#else
  {
    const v8h* s = (const v8h*)src;
    v8h* d = (v8h*)dst;
    for (int c = threadIdx.x; c < chunks16 * 2; c += 256) d[c] = s[c];
  }
#endif
}

// ---------------------------------------------------------------- utilities
__global__ void zero_u32(uint32_t* __restrict__ p, size_t n) {
  size_t i = (size_t)blockIdx.x * blockDim.x + threadIdx.x;
  size_t st = (size_t)gridDim.x * blockDim.x;
  for (; i < n; i += st) p[i] = 0u;
}

// Wc2T[co][k=ci*16+kh*4+kw] (f16): identity-indexed convert of ew2 (OIHW)
__global__ void prep_wc2(const float* __restrict__ ew2, _Float16* __restrict__ Wc2T) {
  int j = blockIdx.x * 256 + threadIdx.x;            // 128*2048
  Wc2T[j] = (_Float16)ew2[j];
}

// Wd1T[cls][co][k=ci*4+t] (f16) from dw1[co][ci][kh][kw]; kh=ph+2*(t>>1), kw=pw+2*(t&1)
__global__ void prep_wd1(const float* __restrict__ dw1, _Float16* __restrict__ Wd1T) {
  int j = blockIdx.x * 256 + threadIdx.x;            // 4*128*512
  int k = j & 511, t2 = j >> 9;
  int co = t2 & 127, cls = t2 >> 7;
  int ph = cls >> 1, pw = cls & 1;
  int ci = k >> 2, t = k & 3;
  int kh = ph + ((t >> 1) << 1), kw = pw + ((t & 1) << 1);
  Wd1T[j] = (_Float16)dw1[((co * 128 + ci) * 4 + kh) * 4 + kw];
}

// Wd2[cls][k=ci*4+t][co(3)] (f32) from dw2[co][ci][kh][kw]
__global__ void prep_wd2(const float* __restrict__ dw2, float* __restrict__ Wd2) {
  int j = blockIdx.x * 256 + threadIdx.x;            // 4*512*3 = 6144
  int co = j % 3, t2 = j / 3;
  int k = t2 & 511, cls = t2 >> 9;
  int ph = cls >> 1, pw = cls & 1;
  int ci = k >> 2, t = k & 3;
  int kh = ph + ((t >> 1) << 1), kw = pw + ((t & 1) << 1);
  Wd2[j] = dw2[((co * 128 + ci) * 4 + kh) * 4 + kw];
}

__global__ void prep_E(const float* __restrict__ E, _Float16* __restrict__ Ef) {
  int j = blockIdx.x * 256 + threadIdx.x;            // 512*128
  Ef[j] = (_Float16)E[j];
}

__global__ void prep_E2(const float* __restrict__ E, float* __restrict__ E2) {
  int c = blockIdx.x * 256 + threadIdx.x;
  if (c >= 512) return;
  float s = 0.f;
  for (int k = 0; k < 128; ++k) { float v = E[c * 128 + k]; s += v * v; }
  E2[c] = s;
}

// ------------------------------------------------- encoder conv1 (VALU, K=48)
// x[32,3,128,128] f32 -> hp[32,128,66,66] f16 (zero halo of 1, ReLU applied)
__global__ __launch_bounds__(256) void conv1_kernel(
    const float* __restrict__ x, const float* __restrict__ w,
    const float* __restrict__ bias, _Float16* __restrict__ hp) {
  int gid = blockIdx.x * 256 + threadIdx.x;          // 32*128*64*64
  int ow = gid & 63, t = gid >> 6;
  int oh = t & 63; t >>= 6;
  int co = t & 127; int b = t >> 7;
  float acc = bias[co];
  for (int ci = 0; ci < 3; ++ci) {
    const float* xp = x + (((b * 3 + ci)) << 14);
    const float* wp = w + ((co * 3 + ci) << 4);
    #pragma unroll
    for (int kh = 0; kh < 4; ++kh) {
      int ih = oh * 2 - 1 + kh;
      if (ih < 0 || ih > 127) continue;
      #pragma unroll
      for (int kw = 0; kw < 4; ++kw) {
        int iw = ow * 2 - 1 + kw;
        if (iw < 0 || iw > 127) continue;
        acc += xp[(ih << 7) + iw] * wp[(kh << 2) + kw];
      }
    }
  }
  acc = acc > 0.f ? acc : 0.f;
  hp[((b * 128 + co) * 66 + (oh + 1)) * 66 + (ow + 1)] = (_Float16)acc;
}

// -------------------------------------- encoder conv2 (WMMA implicit GEMM)
// M=32768 pixels, N=128 cout, K=2048. hp (padded) -> Ze[32768][128] f16
__global__ __launch_bounds__(256) void conv2_wmma(
    const _Float16* __restrict__ hp, const _Float16* __restrict__ Wc2T,
    const float* __restrict__ eb2, _Float16* __restrict__ Ze) {
  __shared__ alignas(32) _Float16 Bsh[16 * 2048];    // [n][k] panel, 64KB, contiguous in global
  int cobase = blockIdx.y << 4;
  stage_panel_256(Wc2T + (size_t)cobase * 2048, Bsh, (16 * 2048 * 2) / 16);
  __syncthreads();

  int wave = threadIdx.x >> 5, lane = threadIdx.x & 31;
  int tilebase = (blockIdx.x << 7) + (wave << 4);
  int m = tilebase + (lane & 15);
  int ow = m & 31, t2 = m >> 5;
  int oh = t2 & 31; int b = t2 >> 5;
  int khalf = (lane & 16) ? 8 : 0;                   // A-frag K split per lane half
  int kh_a = khalf >> 2;                             // 0 or 2
  int n = lane & 15;
  const _Float16* bptr = &Bsh[(n << 11) + ((lane & 16) ? 8 : 0)];
  v8f acc = {};

  for (int k0 = 0; k0 < 2048; k0 += 32) {
    int ci0 = k0 >> 4;
    V16 a;
    #pragma unroll
    for (int g = 0; g < 4; ++g) {                    // (ci0..ci0+1) x (kh_a..kh_a+1)
      int ci = ci0 + (g >> 1);
      int kh = kh_a + (g & 1);
      const uint32_t* p32 = (const uint32_t*)(hp +
          (((b * 128 + ci) * 66 + (oh * 2 + kh)) * 66 + ow * 2));
      G4 g4; g4.u[0] = p32[0]; g4.u[1] = p32[1];     // 4 contiguous taps (kw 0..3)
      #pragma unroll
      for (int jj = 0; jj < 4; ++jj) a.v[g * 4 + jj] = g4.f[jj];
    }
    V16 bf;
    bf.h[0] = *(const v8h*)(bptr + k0);
    bf.h[1] = *(const v8h*)(bptr + k0 + 16);
    acc = wmma_f16(a.v, bf.v, acc);
  }
  float bias = eb2[cobase + n];
  #pragma unroll
  for (int r = 0; r < 8; ++r) {
    int mm = tilebase + r + ((lane & 16) ? 8 : 0);
    Ze[mm * 128 + cobase + n] = (_Float16)(acc[r] + bias);
  }
}

// ------------------------------ VQ: scores = |E_k|^2 - 2*Ze.E^T, argmin (WMMA)
__global__ __launch_bounds__(256) void vq_wmma(
    const _Float16* __restrict__ Ze, const _Float16* __restrict__ Ef,
    const float* __restrict__ E2, int* __restrict__ idx) {
  __shared__ float sc[16 * 512];                     // 32KB score tile
  __shared__ float pmin[16 * 16];
  __shared__ int   pidx[16 * 16];
  int wave = threadIdx.x >> 5, lane = threadIdx.x & 31;
  int mbase = blockIdx.x << 4;
  int mrow = mbase + (lane & 15);
  int khalf = (lane & 16) ? 8 : 0;
  int koff2 = (lane & 16) ? 8 : 0;
  int n = lane & 15;

  V16 a[4];
  #pragma unroll
  for (int s = 0; s < 4; ++s) {                      // K = 128 -> 4 steps of 32
    const _Float16* p = Ze + mrow * 128 + s * 32;
    a[s].h[0] = *(const v8h*)(p + khalf);
    a[s].h[1] = *(const v8h*)(p + 16 + khalf);
  }
  #pragma unroll
  for (int ct = 0; ct < 4; ++ct) {                   // 8 waves x 4 tiles = 512 codes
    int code = (((wave << 2) + ct) << 4) + n;
    const _Float16* ep = Ef + code * 128 + koff2;
    v8f acc = {};
    #pragma unroll
    for (int s = 0; s < 4; ++s) {
      V16 bf;
      bf.h[0] = *(const v8h*)(ep + s * 32);
      bf.h[1] = *(const v8h*)(ep + s * 32 + 16);
      acc = wmma_f16(a[s].v, bf.v, acc);
    }
    float e2 = E2[code];
    #pragma unroll
    for (int r = 0; r < 8; ++r) {
      int mm = r + ((lane & 16) ? 8 : 0);
      sc[mm * 512 + code] = e2 - 2.0f * acc[r];
    }
  }
  __syncthreads();
  // argmin, first-min tie order preserved (ascending scan)
  int row = threadIdx.x >> 4, chunk = threadIdx.x & 15;
  const float* sr = &sc[row * 512 + chunk * 32];
  float best = sr[0]; int bi = chunk * 32;
  for (int j = 1; j < 32; ++j) {
    float v = sr[j];
    if (v < best) { best = v; bi = chunk * 32 + j; }
  }
  pmin[row * 16 + chunk] = best; pidx[row * 16 + chunk] = bi;
  __syncthreads();
  if (threadIdx.x < 16) {
    float bb = pmin[threadIdx.x * 16]; int bj = pidx[threadIdx.x * 16];
    for (int c = 1; c < 16; ++c) {
      float v = pmin[threadIdx.x * 16 + c];
      if (v < bb) { bb = v; bj = pidx[threadIdx.x * 16 + c]; }
    }
    idx[mbase + threadIdx.x] = bj;
  }
}

// dec_in.flat[i] = E[idx[i>>7]][i&127]  (faithful reshape without permute-back)
// scatter into padded dp[32,128,34,34] f16
__global__ __launch_bounds__(256) void scatter_dec(
    const int* __restrict__ idx, const _Float16* __restrict__ Ef,
    _Float16* __restrict__ dp) {
  int i = blockIdx.x * 256 + threadIdx.x;            // 4.19M, NCHW flat
  int r = i & 127, q = i >> 7;
  _Float16 val = Ef[idx[q] * 128 + r];
  int xw = i & 31, t = i >> 5;
  int y = t & 31; t >>= 5;
  int c = t & 127; int b = t >> 7;
  dp[((b * 128 + c) * 34 + (y + 1)) * 34 + (xw + 1)] = val;
}

// ------------------- decoder deconv1 (WMMA, 4 parity classes, K=512 each)
// dp (padded) -> gp[32,128,66,66] f16 (zero halo, ReLU)
__global__ __launch_bounds__(256) void deconv1_wmma(
    const _Float16* __restrict__ dp, const _Float16* __restrict__ Wd1T,
    const float* __restrict__ db1, _Float16* __restrict__ gp) {
  __shared__ alignas(32) _Float16 Bsh[16 * 512];     // [n][k] panel, 16KB, contiguous
  int cls = blockIdx.z; int ph = cls >> 1, pw = cls & 1;
  int cobase = blockIdx.y << 4;
  stage_panel_256(Wd1T + ((size_t)cls * 128 + cobase) * 512, Bsh, (16 * 512 * 2) / 16);
  __syncthreads();

  int wave = threadIdx.x >> 5, lane = threadIdx.x & 31;
  int tilebase = (blockIdx.x << 7) + (wave << 4);
  int m = tilebase + (lane & 15);
  int owp = m & 31, t2 = m >> 5;
  int ohp = t2 & 31; int b = t2 >> 5;
  int r0 = ohp + ph, c0 = owp + pw;                  // padded input base coords
  int khalf = (lane & 16) ? 8 : 0;
  int n = lane & 15;
  const _Float16* bptr = &Bsh[(n << 9) + ((lane & 16) ? 8 : 0)];
  v8f acc = {};

  for (int k0 = 0; k0 < 512; k0 += 32) {             // K = ci*4 + tap
    int cib = (k0 + khalf) >> 2;
    V16 a;
    #pragma unroll
    for (int q = 0; q < 4; ++q) {                    // ci offsets 0,1,4,5
      int ci = cib + ((q & 1) + ((q >> 1) << 2));
      const _Float16* p = dp + ((b * 128 + ci) * 34 + r0) * 34 + c0;
      a.v[q * 4 + 0] = p[0];
      a.v[q * 4 + 1] = p[1];
      a.v[q * 4 + 2] = p[34];
      a.v[q * 4 + 3] = p[35];
    }
    V16 bf;
    bf.h[0] = *(const v8h*)(bptr + k0);
    bf.h[1] = *(const v8h*)(bptr + k0 + 16);
    acc = wmma_f16(a.v, bf.v, acc);
  }
  float bias = db1[cobase + n];
  #pragma unroll
  for (int r = 0; r < 8; ++r) {
    int mm = tilebase + r + ((lane & 16) ? 8 : 0);
    int oww = mm & 31, tt = mm >> 5;
    int ohh = tt & 31; int bb = tt >> 5;
    int oh = 2 * ohh + ph, ow = 2 * oww + pw;
    float v = acc[r] + bias; v = v > 0.f ? v : 0.f;
    gp[((bb * 128 + cobase + n) * 66 + (oh + 1)) * 66 + (ow + 1)] = (_Float16)v;
  }
}

// --------------------- decoder deconv2 (VALU, Cout=3) + sigmoid -> d_out f32
__global__ __launch_bounds__(256) void deconv2_kernel(
    const _Float16* __restrict__ gp, const float* __restrict__ Wd2,
    const float* __restrict__ db2, float* __restrict__ out) {
  int gid = blockIdx.x * 256 + threadIdx.x;          // 32*128*128 pixels
  int ow = gid & 127, t = gid >> 7;
  int oh = t & 127; int b = t >> 7;
  int ph = oh & 1, pw = ow & 1;
  int r0 = (oh >> 1) + ph, c0 = (ow >> 1) + pw;      // padded gp coords
  const float* W = Wd2 + (ph * 2 + pw) * 512 * 3;
  float a0 = db2[0], a1 = db2[1], a2 = db2[2];
  for (int ci = 0; ci < 128; ++ci) {
    const _Float16* p = gp + ((b * 128 + ci) * 66 + r0) * 66 + c0;
    float g0 = (float)p[0], g1 = (float)p[1];
    float g2 = (float)p[66], g3 = (float)p[67];
    const float* w = W + ci * 12;
    a0 += g0 * w[0] + g1 * w[3] + g2 * w[6] + g3 * w[9];
    a1 += g0 * w[1] + g1 * w[4] + g2 * w[7] + g3 * w[10];
    a2 += g0 * w[2] + g1 * w[5] + g2 * w[8] + g3 * w[11];
  }
  int pix = (oh << 7) + ow;
  int ob = b * 3 * 16384;
  out[ob + pix]             = 1.f / (1.f + __expf(-a0));
  out[ob + 16384 + pix]     = 1.f / (1.f + __expf(-a1));
  out[ob + 2 * 16384 + pix] = 1.f / (1.f + __expf(-a2));
}

// ---------------------------------------------------------------------------
extern "C" void kernel_launch(void* const* d_in, const int* in_sizes, int n_in,
                              void* d_out, int out_size, void* d_ws, size_t ws_size,
                              hipStream_t stream) {
  (void)in_sizes; (void)n_in; (void)out_size; (void)ws_size;
  const float* x   = (const float*)d_in[0];
  const float* ew1 = (const float*)d_in[1];
  const float* eb1 = (const float*)d_in[2];
  const float* ew2 = (const float*)d_in[3];
  const float* eb2 = (const float*)d_in[4];
  const float* E   = (const float*)d_in[5];
  const float* dw1 = (const float*)d_in[6];
  const float* db1 = (const float*)d_in[7];
  const float* dw2 = (const float*)d_in[8];
  const float* db2 = (const float*)d_in[9];
  float* out = (float*)d_out;

  char* ws = (char*)d_ws;
  size_t off = 0;
  auto alloc = [&](size_t bytes) {
    char* p = ws + off;
    off = (off + bytes + 255) & ~(size_t)255;
    return p;
  };
  // padded activation buffers first (zeroed together each call)
  _Float16* hp = (_Float16*)alloc((size_t)32 * 128 * 66 * 66 * 2);
  _Float16* dp = (_Float16*)alloc((size_t)32 * 128 * 34 * 34 * 2);
  _Float16* gp = (_Float16*)alloc((size_t)32 * 128 * 66 * 66 * 2);
  size_t zero_bytes = off;
  _Float16* Ze  = (_Float16*)alloc((size_t)32768 * 128 * 2);
  int*      idx = (int*)     alloc((size_t)32768 * 4);
  _Float16* Wc2 = (_Float16*)alloc((size_t)128 * 2048 * 2);
  _Float16* Wd1 = (_Float16*)alloc((size_t)4 * 128 * 512 * 2);
  float*    Wd2 = (float*)   alloc((size_t)4 * 512 * 3 * 4);
  _Float16* Ef  = (_Float16*)alloc((size_t)512 * 128 * 2);
  float*    E2  = (float*)   alloc((size_t)512 * 4);

  zero_u32<<<2048, 256, 0, stream>>>((uint32_t*)ws, zero_bytes / 4);
  prep_wc2<<<262144 / 256, 256, 0, stream>>>(ew2, Wc2);
  prep_wd1<<<262144 / 256, 256, 0, stream>>>(dw1, Wd1);
  prep_wd2<<<24, 256, 0, stream>>>(dw2, Wd2);
  prep_E  <<<65536 / 256, 256, 0, stream>>>(E, Ef);
  prep_E2 <<<2, 256, 0, stream>>>(E, E2);

  conv1_kernel <<<65536, 256, 0, stream>>>(x, ew1, eb1, hp);
  conv2_wmma   <<<dim3(256, 8), 256, 0, stream>>>(hp, Wc2, eb2, Ze);
  vq_wmma      <<<2048, 256, 0, stream>>>(Ze, Ef, E2, idx);
  scatter_dec  <<<16384, 256, 0, stream>>>(idx, Ef, dp);
  deconv1_wmma <<<dim3(256, 8, 4), 256, 0, stream>>>(dp, Wd1, db1, gp);
  deconv2_kernel<<<2048, 256, 0, stream>>>(gp, Wd2, db2, out);
}
